// UAV_55602646614217
// MI455X (gfx1250) — matrix operand order repaired and checked by reading, not compiled, not run on hardware
//
#include <hip/hip_runtime.h>
#include <hip/hip_bf16.h>
#include <math.h>

typedef __attribute__((ext_vector_type(16))) _Float16 v16h;
typedef __attribute__((ext_vector_type(8)))  _Float16 v8h;
typedef __attribute__((ext_vector_type(8)))  float    v8f;

#define ALPHA_ 0.2f
// U=1024, M=64, H=256, Hc=128, N=2112, users rows = 2048, D=2048 edges/UAV

__device__ __forceinline__ float sigf(float x){ return 1.0f/(1.0f+__expf(-x)); }

__device__ __forceinline__ v8f zero8(){
  v8f z;
#pragma unroll
  for (int i=0;i<8;++i) z[i]=0.0f;
  return z;
}

// A fragment: 16x32 f16, row-major source with leading dim lda (halfs).
// Lane L: row m=L&15; halfs[j]=K(hi*16+j), j=0..15
__device__ __forceinline__ v16h load_a_frag(const _Float16* base, int lda){
  int lane = threadIdx.x & 31;
  int m = lane & 15, hi = lane >> 4;
  const _Float16* p = base + m*lda + hi*16;
  v8h lo = *(const v8h*)p;
  v8h hh = *(const v8h*)(p + 8);
  v16h a;
#pragma unroll
  for (int i=0;i<8;++i){ a[i]=lo[i]; a[8+i]=hh[i]; }
  return a;
}

// B fragment: 32x16 (KxN) from Bt stored as [N][K] f16 (ldb halfs).
// Lane L: col n=n0+(L&15); halfs[j] = B[k0 + hi*16 + j][n] = Bt[n][k0+hi*16+j]
__device__ __forceinline__ v16h load_b_frag(const _Float16* bt, int n0, int k0, int ldb){
  int lane = threadIdx.x & 31;
  int n = n0 + (lane & 15), hi = lane >> 4;
  return *(const v16h*)(bt + (size_t)n*ldb + k0 + hi*16);
}

// ---------------- weight conversion (f32 -> f16, split concat halves) ----------------
__global__ void convert_weights_kernel(const float* __restrict__ msg_w1,
                                       const float* __restrict__ msg_w2,
                                       const float* __restrict__ upd_w1,
                                       const float* __restrict__ upd_w2,
                                       _Float16* w1bh, _Float16* w2h,
                                       _Float16* wuah, _Float16* wubh, _Float16* wu2h){
  int i = blockIdx.x*blockDim.x + threadIdx.x;
  if (i >= 65536) return;
  int n = i >> 8, k = i & 255;
  w2h[i]  = (_Float16)msg_w2[i];
  wu2h[i] = (_Float16)upd_w2[i];
  w1bh[i] = (_Float16)msg_w1[n*512 + 256 + k];   // user (xj) half
  wuah[i] = (_Float16)upd_w1[n*512 + k];         // agg half
  wubh[i] = (_Float16)upd_w1[n*512 + 256 + k];   // h half
}

// vr = wr_w^T @ att_r ; consts[0] = att_r . wr_b
__global__ void attvec_kernel(const float* __restrict__ att_w, const float* __restrict__ wr_w,
                              const float* __restrict__ wr_b, float* vr, float* consts){
  int k = threadIdx.x;
  float s = 0.0f;
  for (int j=0;j<256;++j) s += att_w[256+j]*wr_w[j*256+k];
  vr[k] = s;
  __shared__ float red[256];
  red[k] = att_w[256+k]*wr_b[k];
  __syncthreads();
  for (int st=128; st>0; st>>=1){ if (k<st) red[k]+=red[k+st]; __syncthreads(); }
  if (k==0) consts[0] = red[0];
}

// ---------------- users bidirectional LSTM (seq=2, batch=1024) ----------------
__global__ __launch_bounds__(128) void users_lstm_kernel(
    const float* __restrict__ x, const float* __restrict__ wih,
    const float* __restrict__ whh, const float* __restrict__ bb,
    float* __restrict__ h){
  int u = blockIdx.x, t = threadIdx.x;
  float x0a = x[u*2],        x0b = x[u*2+1];
  float x1a = x[(u+1024)*2], x1b = x[(u+1024)*2+1];
  __shared__ float hA[128];
  for (int dir=0; dir<2; ++dir){
    const float* wi = wih + dir*512*2;
    const float* wh = whh + dir*512*128;
    const float* bd = bb  + dir*512;
    float ia = dir ? x1a : x0a, ib = dir ? x1b : x0b;  // first input
    float ja = dir ? x0a : x1a, jb = dir ? x0b : x1b;  // second input
    // step 1 (h0=c0=0)
    float gi = wi[t*2]*ia       + wi[t*2+1]*ib       + bd[t];
    float gg = wi[(256+t)*2]*ia + wi[(256+t)*2+1]*ib + bd[256+t];
    float go = wi[(384+t)*2]*ia + wi[(384+t)*2+1]*ib + bd[384+t];
    float c1 = sigf(gi)*tanhf(gg);
    float h1 = sigf(go)*tanhf(c1);
    hA[t] = h1;
    __syncthreads();
    // step 2
    float di = bd[t]     + wi[t*2]*ja       + wi[t*2+1]*jb;
    float df = bd[128+t] + wi[(128+t)*2]*ja + wi[(128+t)*2+1]*jb;
    float dg = bd[256+t] + wi[(256+t)*2]*ja + wi[(256+t)*2+1]*jb;
    float dO = bd[384+t] + wi[(384+t)*2]*ja + wi[(384+t)*2+1]*jb;
    for (int k=0;k<128;++k){
      float hv = hA[k];
      di += wh[t*128+k]*hv;
      df += wh[(128+t)*128+k]*hv;
      dg += wh[(256+t)*128+k]*hv;
      dO += wh[(384+t)*128+k]*hv;
    }
    float c2  = sigf(df)*c1 + sigf(di)*tanhf(dg);
    float h2v = sigf(dO)*tanhf(c2);
    if (dir==0){                       // forward: out at t=0 is h1, t=1 is h2
      h[u*256 + t]          = h1;
      h[(u+1024)*256 + t]   = h2v;
    } else {                           // reverse: out at t=1 is h1, t=0 is h2
      h[(u+1024)*256 + 128 + t] = h1;
      h[u*256 + 128 + t]        = h2v;
    }
    __syncthreads();
  }
}

// ---------------- UAV linear embedding ----------------
__global__ void uav_embed_kernel(const float* __restrict__ x, const float* __restrict__ w,
                                 const float* __restrict__ b, float* __restrict__ h){
  int i = blockIdx.x*blockDim.x + threadIdx.x;
  if (i >= 64*256) return;
  int m = i >> 8, j = i & 255;
  float xa = x[(2048+m)*2], xb = x[(2048+m)*2+1];
  h[(2048+m)*256 + j] = w[j*2]*xa + w[j*2+1]*xb + b[j];
}

// ---------------- per-UAV: uav_part = uav@W1a^T + b1 ; c0 = q.att_q + att_b + att_r.wr_b ----
__global__ __launch_bounds__(256) void uav_part_kernel(
    const float* __restrict__ h, const float* __restrict__ msg_w1, const float* __restrict__ msg_b1,
    const float* __restrict__ wq_w, const float* __restrict__ wq_b,
    const float* __restrict__ att_w, const float* __restrict__ att_b,
    const float* __restrict__ consts, float* __restrict__ uav_part, float* __restrict__ c0){
  int m = blockIdx.x, t = threadIdx.x;
  __shared__ float hu[256];
  __shared__ float red[256];
  hu[t] = h[(2048+m)*256 + t];
  __syncthreads();
  float s = msg_b1[t];
  const float* w1r = msg_w1 + t*512;          // first 256 cols = xi (UAV) half
  for (int k=0;k<256;++k) s += hu[k]*w1r[k];
  uav_part[m*256 + t] = s;
  float q = wq_b[t];
  const float* wqr = wq_w + t*256;
  for (int k=0;k<256;++k) q += hu[k]*wqr[k];
  red[t] = q * att_w[t];
  __syncthreads();
  for (int st=128; st>0; st>>=1){ if (t<st) red[t]+=red[t+st]; __syncthreads(); }
  if (t==0) c0[m] = red[0] + att_b[0] + consts[0];
}

// ---------------- generic WMMA GEMM: out = act(A[M,256] @ Bt^T + bias + addmat) -----------
__global__ __launch_bounds__(256) void wmma_gemm_kernel(
    const float* __restrict__ A, const _Float16* __restrict__ Bt,
    const float* __restrict__ bias, const float* __restrict__ addmat, int addrow0,
    float* __restrict__ out, int Mrows, int relu){
  __shared__ _Float16 Alds[64*264];
  int row0 = blockIdx.x*64;
  for (int t = threadIdx.x; t < 64*256; t += 256){
    int r = t >> 8, k = t & 255;
    int row = row0 + r;
    float v = (row < Mrows) ? A[(size_t)row*256 + k] : 0.0f;
    Alds[r*264 + k] = (_Float16)v;
  }
  __syncthreads();
  int w = threadIdx.x >> 5, lane = threadIdx.x & 31;
  int hi = lane >> 4, ln = lane & 15;
  v8f cc[4][2];
#pragma unroll
  for (int t=0;t<4;++t){ cc[t][0]=zero8(); cc[t][1]=zero8(); }
  for (int ks=0; ks<8; ++ks){
    int k0 = ks*32;
    v16h a[4];
#pragma unroll
    for (int t=0;t<4;++t) a[t] = load_a_frag(&Alds[t*16*264 + k0], 264);
    v16h b[2];
#pragma unroll
    for (int cs=0;cs<2;++cs) b[cs] = load_b_frag(Bt, w*32 + cs*16, k0, 256);
#pragma unroll
    for (int t=0;t<4;++t)
#pragma unroll
      for (int cs=0;cs<2;++cs)
        cc[t][cs] = __builtin_amdgcn_wmma_f32_16x16x32_f16(
            false, a[t], false, b[cs], (short)0, cc[t][cs], false, false);
  }
#pragma unroll
  for (int t=0;t<4;++t)
#pragma unroll
    for (int cs=0;cs<2;++cs){
      int col = w*32 + cs*16 + ln;
      float bi = bias ? bias[col] : 0.0f;
#pragma unroll
      for (int r=0;r<8;++r){
        int row = row0 + t*16 + r + 8*hi;
        if (row < Mrows){
          float v = cc[t][cs][r] + bi;
          if (addmat && row >= addrow0) v += addmat[(size_t)(row-addrow0)*256 + col];
          if (relu) v = fmaxf(v, 0.0f);
          out[(size_t)row*256 + col] = v;
        }
      }
    }
}

// ------- fused edge kernel, split 4 ways per UAV: msg GEMM + online softmax partials -------
// grid: 256 blocks; block (m = bx>>2, p = bx&3) handles users [p*512, p*512+512)
__global__ __launch_bounds__(256) void edge_attn_kernel(
    const float* __restrict__ user_part, const float* __restrict__ uav_part,
    const _Float16* __restrict__ w2h, const float* __restrict__ vr,
    const float* __restrict__ c0, const float* __restrict__ b2,
    float* __restrict__ part_acc, float* __restrict__ part_mz){
  __shared__ _Float16 Alds[64*264];
  __shared__ float uavp[256];
  __shared__ float s_lds[64];
  __shared__ float w_lds[64];
  __shared__ float red[64];
  __shared__ float agg_lds[256];
  __shared__ float st_max, st_Z, st_scale;

  int m = blockIdx.x >> 2, p = blockIdx.x & 3;
  int tid = threadIdx.x;
  uavp[tid] = uav_part[m*256 + tid];
  agg_lds[tid] = 0.0f;
  if (tid == 0){ st_max = -INFINITY; st_Z = 0.0f; st_scale = 0.0f; }
  float c0m = c0[m];
  int w = tid >> 5, lane = tid & 31, hi = lane >> 4, ln = lane & 15;
  float vrc[2], b2c[2];
#pragma unroll
  for (int cs=0;cs<2;++cs){ int col = w*32+cs*16+ln; vrc[cs]=vr[col]; b2c[cs]=b2[col]; }
  float acc[2] = {0.0f, 0.0f};
  __syncthreads();

  for (int tile=0; tile<8; ++tile){
    int u0 = (p*8 + tile)*64;
    // A = relu(user_part + uav_part(+b1)) -> f16 LDS
    for (int t = tid; t < 64*256; t += 256){
      int r = t >> 8, k = t & 255;
      float v = user_part[(size_t)(u0 + r)*256 + k] + uavp[k];
      Alds[r*264 + k] = (_Float16)fmaxf(v, 0.0f);
    }
    // prefetch next tile of user_part toward this WGP while WMMAs run
    // (lowers to global_prefetch_b8; 512 x 128B lines = 64KB)
    if (tile + 1 < 8){
      const float* np = user_part + (size_t)(u0 + 64)*256;
      for (int j = tid; j < 512; j += 256)
        __builtin_prefetch(np + j*32, 0, 0);
    }
    if (tid < 64) s_lds[tid] = 0.0f;
    __syncthreads();

    v8f cc[4][2];
#pragma unroll
    for (int t=0;t<4;++t){ cc[t][0]=zero8(); cc[t][1]=zero8(); }
    for (int ks=0; ks<8; ++ks){
      int k0 = ks*32;
      v16h a[4];
#pragma unroll
      for (int t=0;t<4;++t) a[t] = load_a_frag(&Alds[t*16*264 + k0], 264);
      v16h b[2];
#pragma unroll
      for (int cs=0;cs<2;++cs) b[cs] = load_b_frag(w2h, w*32 + cs*16, k0, 256);
#pragma unroll
      for (int t=0;t<4;++t)
#pragma unroll
        for (int cs=0;cs<2;++cs)
          cc[t][cs] = __builtin_amdgcn_wmma_f32_16x16x32_f16(
              false, a[t], false, b[cs], (short)0, cc[t][cs], false, false);
    }
    // bias + relu -> msg in regs; partial scores s[row] += msg * vr[col]
#pragma unroll
    for (int t=0;t<4;++t)
#pragma unroll
      for (int cs=0;cs<2;++cs)
#pragma unroll
        for (int r=0;r<8;++r){
          float v = fmaxf(cc[t][cs][r] + b2c[cs], 0.0f);
          cc[t][cs][r] = v;
          atomicAdd(&s_lds[t*16 + r + 8*hi], v * vrc[cs]);
        }
    __syncthreads();
    // online softmax (parallel over 64 rows)
    if (tid < 64){
      float sv = s_lds[tid] + c0m;
      sv = sv > 0.0f ? sv : ALPHA_*sv;          // leaky_relu
      s_lds[tid] = sv;
      red[tid] = sv;
    }
    __syncthreads();
    for (int st=32; st>0; st>>=1){ if (tid<st) red[tid]=fmaxf(red[tid],red[tid+st]); __syncthreads(); }
    if (tid == 0){
      float nm = fmaxf(st_max, red[0]);
      st_scale = __expf(st_max - nm);
      st_max = nm;
    }
    __syncthreads();
    if (tid < 64){
      float wv = __expf(s_lds[tid] - st_max);
      w_lds[tid] = wv;
      red[tid] = wv;
    }
    __syncthreads();
    for (int st=32; st>0; st>>=1){ if (tid<st) red[tid]+=red[tid+st]; __syncthreads(); }
    if (tid == 0) st_Z = st_Z*st_scale + red[0];
    __syncthreads();
    float sc = st_scale;
#pragma unroll
    for (int cs=0;cs<2;++cs){
      float a2 = acc[cs]*sc;
#pragma unroll
      for (int t=0;t<4;++t)
#pragma unroll
        for (int r=0;r<8;++r)
          a2 += w_lds[t*16 + r + 8*hi] * cc[t][cs][r];
      acc[cs] = a2;
    }
    __syncthreads();
  }
#pragma unroll
  for (int cs=0;cs<2;++cs) atomicAdd(&agg_lds[w*32+cs*16+ln], acc[cs]);
  __syncthreads();
  part_acc[(size_t)(m*4+p)*256 + tid] = agg_lds[tid];    // unnormalized, relative to st_max
  if (tid == 0){
    part_mz[(m*4+p)*2 + 0] = st_max;
    part_mz[(m*4+p)*2 + 1] = st_Z;
  }
}

// combine 4 softmax partials per UAV: agg = (sum_p acc_p*e^(m_p-gm)) / (Z * 2048)
__global__ __launch_bounds__(256) void edge_combine_kernel(
    const float* __restrict__ part_acc, const float* __restrict__ part_mz,
    float* __restrict__ agg){
  int m = blockIdx.x, tid = threadIdx.x;
  __shared__ float pm[4], pz[4];
  if (tid < 4){
    pm[tid] = part_mz[(m*4+tid)*2 + 0];
    pz[tid] = part_mz[(m*4+tid)*2 + 1];
  }
  __syncthreads();
  float gm = fmaxf(fmaxf(pm[0],pm[1]), fmaxf(pm[2],pm[3]));
  float Z = 0.0f, s = 0.0f;
#pragma unroll
  for (int p=0;p<4;++p){
    float e = __expf(pm[p] - gm);
    Z += pz[p]*e;
    s += part_acc[(size_t)(m*4+p)*256 + tid]*e;
  }
  agg[m*256 + tid] = s / (Z * 2048.0f);
}

// ---------------- UAV bidirectional LSTM scan over the 64 UAVs (T=64, B=1) ----------------
__global__ __launch_bounds__(128) void uav_lstm_kernel(
    const float* __restrict__ h2, const float* __restrict__ wih,
    const float* __restrict__ whh, const float* __restrict__ bb,
    float* __restrict__ uav_out){
  int dir = blockIdx.x, t = threadIdx.x;
  const float* wi = wih + dir*512*256;
  const float* wh = whh + dir*512*128;
  const float* bd = bb  + dir*512;
  __shared__ float hS[128];
  __shared__ float xS[256];
  hS[t] = 0.0f;
  float c = 0.0f;
  for (int step=0; step<64; ++step){
    int m = dir ? (63-step) : step;
    __syncthreads();
    xS[t]     = h2[(2048+m)*256 + t];
    xS[128+t] = h2[(2048+m)*256 + 128 + t];
    __syncthreads();
    float gi=bd[t], gf=bd[128+t], gg=bd[256+t], go=bd[384+t];
    for (int k=0;k<256;++k){
      float xv = xS[k];
      gi += wi[t*256+k]*xv;
      gf += wi[(128+t)*256+k]*xv;
      gg += wi[(256+t)*256+k]*xv;
      go += wi[(384+t)*256+k]*xv;
    }
    for (int k=0;k<128;++k){
      float hv = hS[k];
      gi += wh[t*128+k]*hv;
      gf += wh[(128+t)*128+k]*hv;
      gg += wh[(256+t)*128+k]*hv;
      go += wh[(384+t)*128+k]*hv;
    }
    c = sigf(gf)*c + sigf(gi)*tanhf(gg);
    float hv = sigf(go)*tanhf(c);
    __syncthreads();
    hS[t] = hv;
    uav_out[m*256 + dir*128 + t] = hv;
  }
}

// ---------------- final: sigmoid(row @ lin2^T + b) ----------------
__global__ void final_kernel(const float* __restrict__ h2, const float* __restrict__ uav_out,
                             const float* __restrict__ lin2_w, const float* __restrict__ lin2_b,
                             float* __restrict__ out){
  int i = blockIdx.x*blockDim.x + threadIdx.x;
  if (i >= 2112) return;
  const float* row = (i < 2048) ? (h2 + (size_t)i*256) : (uav_out + (size_t)(i-2048)*256);
  float s0 = lin2_b[0], s1 = lin2_b[1];
  for (int k=0;k<256;++k){
    float v = row[k];
    s0 += lin2_w[k]*v;
    s1 += lin2_w[256+k]*v;
  }
  out[i*2]   = sigf(s0);
  out[i*2+1] = sigf(s1);
}

extern "C" void kernel_launch(void* const* d_in, const int* in_sizes, int n_in,
                              void* d_out, int out_size, void* d_ws, size_t ws_size,
                              hipStream_t stream){
  (void)in_sizes; (void)n_in; (void)out_size; (void)ws_size;
  const float* x         = (const float*)d_in[0];
  const float* ul_wih    = (const float*)d_in[3];
  const float* ul_whh    = (const float*)d_in[4];
  const float* ul_b      = (const float*)d_in[5];
  const float* uav_lin_w = (const float*)d_in[6];
  const float* uav_lin_b = (const float*)d_in[7];
  const float* msg_w1    = (const float*)d_in[8];
  const float* msg_b1    = (const float*)d_in[9];
  const float* msg_w2    = (const float*)d_in[10];
  const float* msg_b2    = (const float*)d_in[11];
  const float* wq_w      = (const float*)d_in[12];
  const float* wq_b      = (const float*)d_in[13];
  const float* wr_w      = (const float*)d_in[14];
  const float* wr_b      = (const float*)d_in[15];
  const float* att_w     = (const float*)d_in[16];
  const float* att_b     = (const float*)d_in[17];
  const float* upd_w1    = (const float*)d_in[18];
  const float* upd_b1    = (const float*)d_in[19];
  const float* upd_w2    = (const float*)d_in[20];
  const float* upd_b2    = (const float*)d_in[21];
  const float* uavl_wih  = (const float*)d_in[22];
  const float* uavl_whh  = (const float*)d_in[23];
  const float* uavl_b    = (const float*)d_in[24];
  const float* lin2_w    = (const float*)d_in[25];
  const float* lin2_b    = (const float*)d_in[26];

  char* ws = (char*)d_ws;
  size_t off = 0;
  auto alloc = [&](size_t bytes)->char*{
    char* ptr = ws + off;
    off = (off + bytes + 255) & ~(size_t)255;
    return ptr;
  };
  float* h_nodes   = (float*)alloc((size_t)2112*256*4);
  float* user_part = (float*)alloc((size_t)2048*256*4);
  float* uav_part  = (float*)alloc((size_t)64*256*4);
  float* agg       = (float*)alloc((size_t)64*256*4);
  float* aggW      = (float*)alloc((size_t)64*256*4);
  float* t1        = (float*)alloc((size_t)2112*256*4);
  float* h2        = (float*)alloc((size_t)2112*256*4);
  float* uav_out   = (float*)alloc((size_t)64*256*4);
  float* part_acc  = (float*)alloc((size_t)64*4*256*4);
  float* part_mz   = (float*)alloc((size_t)64*4*2*4);
  float* vr        = (float*)alloc(256*4);
  float* c0        = (float*)alloc(64*4);
  float* consts    = (float*)alloc(256*4);
  _Float16* w1bh   = (_Float16*)alloc((size_t)65536*2);
  _Float16* w2h    = (_Float16*)alloc((size_t)65536*2);
  _Float16* wuah   = (_Float16*)alloc((size_t)65536*2);
  _Float16* wubh   = (_Float16*)alloc((size_t)65536*2);
  _Float16* wu2h   = (_Float16*)alloc((size_t)65536*2);

  convert_weights_kernel<<<256,256,0,stream>>>(msg_w1, msg_w2, upd_w1, upd_w2,
                                               w1bh, w2h, wuah, wubh, wu2h);
  attvec_kernel<<<1,256,0,stream>>>(att_w, wr_w, wr_b, vr, consts);
  users_lstm_kernel<<<1024,128,0,stream>>>(x, ul_wih, ul_whh, ul_b, h_nodes);
  uav_embed_kernel<<<64,256,0,stream>>>(x, uav_lin_w, uav_lin_b, h_nodes);
  uav_part_kernel<<<64,256,0,stream>>>(h_nodes, msg_w1, msg_b1, wq_w, wq_b,
                                       att_w, att_b, consts, uav_part, c0);
  // user_part = users @ W1b^T   (2048x256 @ 256x256, WMMA)
  wmma_gemm_kernel<<<32,256,0,stream>>>(h_nodes, w1bh, nullptr, nullptr, 0,
                                        user_part, 2048, 0);
  // fused edge message MLP + attention, split 4 partials per UAV (256 blocks), then combine
  edge_attn_kernel<<<256,256,0,stream>>>(user_part, uav_part, w2h, vr, c0, msg_b2,
                                         part_acc, part_mz);
  edge_combine_kernel<<<64,256,0,stream>>>(part_acc, part_mz, agg);
  // update MLP: aggW = agg @ Wua^T ; t1 = relu(h@Wub^T + b1 + aggW[uav]) ; h2 = relu(t1@Wu2^T + b2)
  wmma_gemm_kernel<<<1,256,0,stream>>>(agg, wuah, nullptr, nullptr, 0, aggW, 64, 0);
  wmma_gemm_kernel<<<33,256,0,stream>>>(h_nodes, wubh, upd_b1, aggW, 2048, t1, 2112, 1);
  wmma_gemm_kernel<<<33,256,0,stream>>>(t1, wu2h, upd_b2, nullptr, 0, h2, 2112, 1);
  uav_lstm_kernel<<<2,128,0,stream>>>(h2, uavl_wih, uavl_whh, uavl_b, uav_out);
  final_kernel<<<9,256,0,stream>>>(h2, uav_out, lin2_w, lin2_b, (float*)d_out);
}
